// MinkEncConvNeXtV2_5360119185825
// MI455X (gfx1250) — matrix operand order repaired and checked by reading, not compiled, not run on hardware
//
#include <hip/hip_runtime.h>
#include <hip/hip_bf16.h>

// ---------------------------------------------------------------------------
// MI455X (gfx1250, wave32) module/event transformer.
// GEMMs: v_wmma_f32_16x16x32_f16 with fragment-packed global B operands.
// Attention staging uses GLOBAL_LOAD_ASYNC_TO_LDS_B128 when available.
// ---------------------------------------------------------------------------

typedef __attribute__((ext_vector_type(16))) _Float16 v16h;
typedef __attribute__((ext_vector_type(8)))  _Float16 v8h;
typedef __attribute__((ext_vector_type(8)))  float    v8f;

#define LN_EPS 1e-5f
#define INV_SQRT_DH 0.17677669529663687f // 1/sqrt(32)

#if defined(__gfx1250__) && __has_builtin(__builtin_amdgcn_global_load_async_to_lds_b128) && \
    __has_builtin(__builtin_amdgcn_s_wait_asynccnt)
#define HAVE_ASYNC_LDS 1
#endif

// copy 8 halfs (16B) global -> LDS
__device__ inline void async_cp16(const _Float16* g, _Float16* l) {
#ifdef HAVE_ASYNC_LDS
  typedef int v4i __attribute__((vector_size(16)));
  typedef __attribute__((address_space(1))) v4i* gp_t;
  typedef __attribute__((address_space(3))) v4i* lp_t;
  __builtin_amdgcn_global_load_async_to_lds_b128((gp_t)(void*)g, (lp_t)(void*)l, 0, 0);
#else
  *(v8h*)l = *(const v8h*)g;
#endif
}
__device__ inline void async_wait() {
#ifdef HAVE_ASYNC_LDS
  __builtin_amdgcn_s_wait_asynccnt(0);
#endif
}

__device__ inline v8f wmma32(v16h a, v16h b, v8f c) {
  // (neg_a, A, neg_b, B, c_mod, C, reuse_a, reuse_b)
  return __builtin_amdgcn_wmma_f32_16x16x32_f16(false, a, false, b, (short)0, c,
                                                false, false);
}

// A fragment: 16 rows x 32 K from LDS row-major (ld halfs) -> ds_load_b128 x2.
__device__ inline v16h ldA(const _Float16* s, int ld) {
  int lane = threadIdx.x & 31;
  const _Float16* p = s + (lane & 15) * ld + ((lane >> 4) << 4);
  v16h a;
#pragma unroll
  for (int e = 0; e < 8; ++e) a[e] = p[e];
#pragma unroll
  for (int e = 0; e < 8; ++e) a[8 + e] = p[8 + e];
  return a;
}

// B fragment from fragment-packed global weights: 32 contiguous bytes/lane.
__device__ inline v16h ldBg(const _Float16* __restrict__ Wp, int kt, int nt, int ntiles) {
  int lane = threadIdx.x & 31;
  const _Float16* p = Wp + ((((size_t)kt * ntiles + nt) * 32 + lane) << 4);
  v16h b;
#pragma unroll
  for (int e = 0; e < 16; ++e) b[e] = p[e];
  return b;
}

// B fragment where source rows are the WMMA N dim and cols are K
// (row-major [n][k], ld halfs): contiguous 16 halfs/lane.
__device__ inline v16h ldBT(const _Float16* s, int n0, int ld) {
  int lane = threadIdx.x & 31;
  const _Float16* p = s + (n0 + (lane & 15)) * ld + ((lane >> 4) << 4);
  v16h b;
#pragma unroll
  for (int e = 0; e < 16; ++e) b[e] = p[e];
  return b;
}

__device__ inline float gelu_exact(float x) {
  return 0.5f * x * (1.f + erff(x * 0.70710678118654752f));
}
__device__ inline float wave_sum(float v) {
#pragma unroll
  for (int o = 16; o > 0; o >>= 1) v += __shfl_xor(v, o);
  return v;
}
__device__ inline float wave_max(float v) {
#pragma unroll
  for (int o = 16; o > 0; o >>= 1) v = fmaxf(v, __shfl_xor(v, o));
  return v;
}

// ------------------------------ setup kernels ------------------------------

__global__ void zero_kernel(float* p, size_t n) {
  size_t i = (size_t)blockIdx.x * blockDim.x + threadIdx.x;
  size_t st = (size_t)gridDim.x * blockDim.x;
  for (; i < n; i += st) p[i] = 0.f;
}

// (L,4,256,256) attention weights -> fragment-tiled f16 [L][kt=8][nt=64][lane=32][16]
__global__ void packqkvo_kernel(const float* __restrict__ w, _Float16* __restrict__ dst, int n) {
  int i = blockIdx.x * 256 + threadIdx.x;
  if (i >= n) return;
  int l = i >> 18;          // 262144 per layer
  int r = i & 262143;
  int tile = r >> 9, inner = r & 511;
  int lane = inner >> 4, e = inner & 15;
  int kt = tile >> 6, nt = tile & 63;
  int k = kt * 32 + ((lane >> 4) << 4) + e;
  int ncol = nt * 16 + (lane & 15);
  int wi = ncol >> 8, j = ncol & 255;
  dst[i] = (_Float16)w[(((l * 4 + wi) * 256 + k) << 8) + j];
}

// generic [L][K][N] f32 -> fragment-tiled f16
__global__ void packw_kernel(const float* __restrict__ w, _Float16* __restrict__ dst,
                             int K, int N, int total) {
  int i = blockIdx.x * 256 + threadIdx.x;
  if (i >= total) return;
  int KN = K * N;
  int l = i / KN, r = i % KN;
  int tile = r >> 9, inner = r & 511;
  int lane = inner >> 4, e = inner & 15;
  int ntiles = N >> 4;
  int kt = tile / ntiles, nt = tile % ntiles;
  int k = kt * 32 + ((lane >> 4) << 4) + e;
  int ncol = nt * 16 + (lane & 15);
  dst[i] = (_Float16)w[(size_t)l * KN + (size_t)k * N + ncol];
}

__global__ void count_kernel(const int* __restrict__ bids, int* counts, int N) {
  int i = blockIdx.x * 256 + threadIdx.x;
  if (i < N) atomicAdd(&counts[bids[i]], 1);
}

__global__ void starts_kernel(const int* __restrict__ counts, int* starts, int M) {
  if (threadIdx.x == 0 && blockIdx.x == 0) {
    int s = 0;
    for (int i = 0; i < M; ++i) { starts[i] = s; s += counts[i]; }
  }
}

__global__ void clsfill_kernel(const float* __restrict__ cls, float* __restrict__ X) {
  X[(size_t)blockIdx.x * 112 * 256 + threadIdx.x] = cls[threadIdx.x];
}

__global__ void scatter_kernel(const float* __restrict__ feats, const float* __restrict__ coords,
                               const int* __restrict__ bids, const int* __restrict__ starts,
                               float* __restrict__ X, float* __restrict__ pc) {
  int i = blockIdx.x, d = threadIdx.x;
  int b = bids[i];
  int pos = i - starts[b];
  X[((size_t)b * 112 + 1 + pos) * 256 + d] = feats[(size_t)i * 256 + d];
  if (d < 3) pc[((size_t)b * 112 + 1 + pos) * 3 + d] = coords[(size_t)i * 3 + d];
}

// ------------------------------- GEMM kernels ------------------------------

// QK[m][S][512] (Q|K row-major) and VT[m][h][32][S] (V transposed) =
// X @ Wqkv + b. grid(nseq, SP/16, 3); block 256; 2 n-tiles per wave.
__global__ void qkv_kernel(const float* __restrict__ X, const _Float16* __restrict__ Wp,
                           const float* __restrict__ bias, _Float16* __restrict__ QK,
                           _Float16* __restrict__ VT, int SP) {
  __shared__ _Float16 a_s[16 * 256];
  int m = blockIdx.x, rb = blockIdx.y, cb = blockIdx.z;
  int t = threadIdx.x, w = t >> 5, lane = t & 31;
  const float* Xrow = X + ((size_t)m * SP + rb * 16) * 256;
  {
    int r = t >> 4, c0 = (t & 15) * 16;
#pragma unroll
    for (int j = 0; j < 16; ++j) a_s[r * 256 + c0 + j] = (_Float16)Xrow[r * 256 + c0 + j];
  }
  __syncthreads();
  int nt0 = cb * 16 + w;
  v8f acc[2] = {};
#pragma unroll
  for (int ks = 0; ks < 8; ++ks) {
    v16h a = ldA(a_s + ks * 32, 256);
    acc[0] = wmma32(a, ldBg(Wp, ks, nt0, 64), acc[0]);
    acc[1] = wmma32(a, ldBg(Wp, ks, nt0 + 8, 64), acc[1]);
  }
  int n = lane & 15, mo = (lane >> 4) * 8;
#pragma unroll
  for (int half = 0; half < 2; ++half) {
    int col = (nt0 + half * 8) * 16 + n;
    float bv = bias[col];
    if (col < 512) {
      _Float16* dst = QK + ((size_t)m * SP + rb * 16 + mo) * 512 + col;
#pragma unroll
      for (int r = 0; r < 8; ++r) dst[(size_t)r * 512] = (_Float16)(acc[half][r] + bv);
    } else {
      int vc = col - 512, h = vc >> 5, c = vc & 31;
      v8h tmp;
#pragma unroll
      for (int r = 0; r < 8; ++r) tmp[r] = (_Float16)(acc[half][r] + bv);
      *(v8h*)(VT + (((size_t)m * 8 + h) * 32 + c) * SP + rb * 16 + mo) = tmp;
    }
  }
}

// Per-(seq, head, 16-row block) attention. pc/counts/rel NULL => event mode
// with fixed key limit. grid(nseq, 8, SP/16); block 256.
__global__ void attn_kernel(const _Float16* __restrict__ QK, const _Float16* __restrict__ VT,
                            _Float16* __restrict__ OH,
                            const float* __restrict__ pc, const int* __restrict__ counts,
                            const float* __restrict__ rel, int SP, int klimit) {
  __shared__ _Float16 q_s[16 * 32];
  __shared__ _Float16 k_s[128 * 32];
  __shared__ _Float16 v_t[32 * 128];   // V^T slice [dh][key], zero-padded keys
  __shared__ float    lg[16 * 128];
  __shared__ _Float16 p_s[16 * 128];
  __shared__ float    cq[16 * 3];
  __shared__ float    ck[128 * 3];
  int m = blockIdx.x, h = blockIdx.y, rb = blockIdx.z;
  int t = threadIdx.x, w = t >> 5, lane = t & 31;
  int KP = (SP + 31) & ~31;  // 128 or 32
  int KT = SP >> 4;          // 7 or 2
  const _Float16* baseqk = QK + (size_t)m * SP * 512;
  v8h zv = {};
  { // Q rows rb*16..+16, head slice: 64 16B chunks
    if (t < 64) {
      int r = t >> 2, c0 = (t & 3) * 8;
      async_cp16(&baseqk[(size_t)(rb * 16 + r) * 512 + h * 32 + c0], &q_s[r * 32 + c0]);
    }
  }
  for (int i = t; i < KP * 4; i += 256) { // K rows: 16B chunks, zero-pad r>=SP
    int r = i >> 2, c0 = (i & 3) * 8;
    if (r < SP)
      async_cp16(&baseqk[(size_t)r * 512 + 256 + h * 32 + c0], &k_s[r * 32 + c0]);
    else
      *(v8h*)&k_s[r * 32 + c0] = zv;
  }
  { // V^T slice: contiguous chunks, zero-pad keys SP..KP
    const _Float16* vsrc = VT + ((size_t)m * 8 + h) * 32 * SP;
    int cpr = KP >> 3;       // 8-half chunks per dh-row
    for (int i = t; i < 32 * cpr; i += 256) {
      int row = i / cpr, kc = (i % cpr) * 8;
      if (kc < SP)
        async_cp16(vsrc + (size_t)row * SP + kc, v_t + row * 128 + kc);
      else
        *(v8h*)&v_t[row * 128 + kc] = zv;
    }
  }
  float r0 = 0.f, r1 = 0.f, r2 = 0.f;
  if (pc) {
    if (t < 48) cq[t] = pc[((size_t)m * SP + rb * 16 + t / 3) * 3 + (t % 3)];
    for (int i = t; i < SP * 3; i += 256) ck[i] = pc[((size_t)m * SP + i / 3) * 3 + (i % 3)];
    r0 = rel[h * 3 + 0]; r1 = rel[h * 3 + 1]; r2 = rel[h * 3 + 2];
  }
  int cnt = counts ? counts[m] : 0;
  async_wait();
  __syncthreads();

  if (w < KT) { // logits tile: 16 queries x 16 keys, K-dim 32 -> ONE wmma
    v16h a = ldA(q_s, 32);
    v16h b = ldBT(k_s, w * 16, 32);
    v8f c = {};
    c = wmma32(a, b, c);
    int n = lane & 15, mo = (lane >> 4) * 8;
    int j = w * 16 + n;
#pragma unroll
    for (int r = 0; r < 8; ++r) {
      int qr = mo + r;
      float val = c[r] * INV_SQRT_DH;
      if (pc) {
        float dx = cq[qr * 3 + 0] - ck[j * 3 + 0];
        float dy = cq[qr * 3 + 1] - ck[j * 3 + 1];
        float dz = cq[qr * 3 + 2] - ck[j * 3 + 2];
        val += dx * r0 + dy * r1 + dz * r2;
      }
      bool mask = counts ? (j > 0 && (j - 1) >= cnt) : (j >= klimit);
      lg[qr * 128 + j] = mask ? -1e9f : val;
    }
  }
  __syncthreads();

  // softmax: each wave owns 2 rows
#pragma unroll
  for (int rr = 0; rr < 2; ++rr) {
    int row = w * 2 + rr;
    float mx = -3e38f;
    for (int j = lane; j < SP; j += 32) mx = fmaxf(mx, lg[row * 128 + j]);
    mx = wave_max(mx);
    float sm = 0.f;
    for (int j = lane; j < SP; j += 32) {
      float e = __expf(lg[row * 128 + j] - mx);
      lg[row * 128 + j] = e;
      sm += e;
    }
    sm = wave_sum(sm);
    float inv = 1.f / sm;
    for (int j = lane; j < KP; j += 32)
      p_s[row * 128 + j] = (j < SP) ? (_Float16)(lg[row * 128 + j] * inv) : (_Float16)0.f;
  }
  __syncthreads();

  if (w < 2) { // O = P @ V : 16x16 per wave, K = KP
    v8f o = {};
    for (int ks = 0; ks < (KP >> 5); ++ks) {
      v16h a = ldA(p_s + ks * 32, 128);
      v16h b = ldBT(v_t + ks * 32, w * 16, 128);  // V^T rows = dh, cols = key
      o = wmma32(a, b, o);
    }
    int n = lane & 15, mo = (lane >> 4) * 8;
    _Float16* dst = OH + ((size_t)m * SP + rb * 16) * 256 + h * 32 + w * 16 + n;
#pragma unroll
    for (int r = 0; r < 8; ++r) dst[(size_t)(mo + r) * 256] = (_Float16)o[r];
  }
}

// X = LN(OH @ Wo + bo + X) ; Wo = nt tiles 48..63 of packed wqkvo.
// grid(nseq, SP/16); block 256.
__global__ void oproj_ln_kernel(const _Float16* __restrict__ OH, const _Float16* __restrict__ Wp,
                                const float* __restrict__ bo, const float* __restrict__ lns,
                                const float* __restrict__ lnb, float* __restrict__ X, int SP) {
  __shared__ _Float16 a_s[16 * 256];
  __shared__ float    outb[16 * 256];
  int m = blockIdx.x, rb = blockIdx.y;
  int t = threadIdx.x, w = t >> 5, lane = t & 31;
  size_t grow0 = (size_t)m * SP + rb * 16;
  for (int i = t; i < 512; i += 256) { // 16x256 A panel, 16B async chunks
    int r = i >> 5, c0 = (i & 31) * 8;
    async_cp16(OH + (grow0 + r) * 256 + c0, a_s + r * 256 + c0);
  }
  async_wait();
  __syncthreads();
  v8f acc[2] = {};
#pragma unroll
  for (int ks = 0; ks < 8; ++ks) {
    v16h a = ldA(a_s + ks * 32, 256);
    acc[0] = wmma32(a, ldBg(Wp, ks, 48 + w, 64), acc[0]);
    acc[1] = wmma32(a, ldBg(Wp, ks, 56 + w, 64), acc[1]);
  }
  int n = lane & 15, mo = (lane >> 4) * 8;
#pragma unroll
  for (int half = 0; half < 2; ++half) {
    int col = half * 128 + w * 16 + n;
#pragma unroll
    for (int r = 0; r < 8; ++r)
      outb[(mo + r) * 256 + col] = acc[half][r] + bo[col] + X[(grow0 + mo + r) * 256 + col];
  }
  __syncthreads();
#pragma unroll
  for (int rr = 0; rr < 2; ++rr) {
    int row = w * 2 + rr;
    const float* pr = outb + row * 256;
    int c0 = lane * 8;
    float s = 0.f;
#pragma unroll
    for (int j = 0; j < 8; ++j) s += pr[c0 + j];
    float mu = wave_sum(s) * (1.f / 256.f);
    float v2 = 0.f;
#pragma unroll
    for (int j = 0; j < 8; ++j) { float d = pr[c0 + j] - mu; v2 += d * d; }
    float inv = rsqrtf(wave_sum(v2) * (1.f / 256.f) + LN_EPS);
    float* xr = X + (grow0 + row) * 256;
#pragma unroll
    for (int j = 0; j < 8; ++j) {
      int c = c0 + j;
      xr[c] = (pr[c] - mu) * inv * lns[c] + lnb[c];
    }
  }
}

// X = LN(act(X@W1+b1)@W2 + b2 + X). N1 in 1024-wide chunks (hidden tile in
// LDS). act: 0=gelu(exact), 1=relu. grid(nseq, SP/16); block 256.
__global__ void ffn_kernel(float* __restrict__ X, const _Float16* __restrict__ W1p,
                           const float* __restrict__ b1, const _Float16* __restrict__ W2p,
                           const float* __restrict__ b2, const float* __restrict__ lns,
                           const float* __restrict__ lnb, int SP, int N1, int act) {
  __shared__ _Float16 xa[16 * 256];
  __shared__ _Float16 h1[16 * 1024];
  int m = blockIdx.x, rb = blockIdx.y;
  int t = threadIdx.x, w = t >> 5, lane = t & 31;
  size_t grow0 = (size_t)m * SP + rb * 16;
  {
    int r = t >> 4, c0 = (t & 15) * 16;
    const float* src = X + (grow0 + r) * 256 + c0;
#pragma unroll
    for (int j = 0; j < 16; ++j) xa[r * 256 + c0 + j] = (_Float16)src[j];
  }
  __syncthreads();
  int ntiles1 = N1 >> 4;
  int n = lane & 15, mo = (lane >> 4) * 8;
  v8f acc[2] = {};
  int nchunk = N1 >> 10;
  for (int ch = 0; ch < nchunk; ++ch) {
    __syncthreads();  // previous chunk's GEMM2 reads of h1 complete
    // GEMM1: h1[16][1024] = act(X @ W1_chunk + b1); 2 n-tiles per wave
    for (int cb = 0; cb < 4; ++cb) {
      v8f c1[2] = {};
      int ntb = ch * 64 + cb * 16 + w;
#pragma unroll
      for (int ks = 0; ks < 8; ++ks) {
        v16h a = ldA(xa + ks * 32, 256);
        c1[0] = wmma32(a, ldBg(W1p, ks, ntb, ntiles1), c1[0]);
        c1[1] = wmma32(a, ldBg(W1p, ks, ntb + 8, ntiles1), c1[1]);
      }
#pragma unroll
      for (int half = 0; half < 2; ++half) {
        int colc = cb * 256 + half * 128 + w * 16 + n;
        int col = ch * 1024 + colc;
#pragma unroll
        for (int r = 0; r < 8; ++r) {
          float v = c1[half][r] + b1[col];
          v = act ? fmaxf(v, 0.f) : gelu_exact(v);
          h1[(mo + r) * 1024 + colc] = (_Float16)v;
        }
      }
    }
    __syncthreads();
    // GEMM2: accumulate h1 @ W2_chunk into register fragments
    for (int ks = 0; ks < 32; ++ks) {
      v16h a = ldA(h1 + ks * 32, 1024);
      int kt = ch * 32 + ks;
      acc[0] = wmma32(a, ldBg(W2p, kt, w, 16), acc[0]);
      acc[1] = wmma32(a, ldBg(W2p, kt, 8 + w, 16), acc[1]);
    }
  }
  __syncthreads();
  // epilogue: residual + LN (reuse h1 region as f32 scratch)
  float* outb = (float*)h1;
#pragma unroll
  for (int half = 0; half < 2; ++half) {
    int col = half * 128 + w * 16 + n;
#pragma unroll
    for (int r = 0; r < 8; ++r)
      outb[(mo + r) * 256 + col] = acc[half][r] + b2[col] + X[(grow0 + mo + r) * 256 + col];
  }
  __syncthreads();
#pragma unroll
  for (int rr = 0; rr < 2; ++rr) {
    int row = w * 2 + rr;
    const float* pr = outb + row * 256;
    int c0 = lane * 8;
    float s = 0.f;
#pragma unroll
    for (int j = 0; j < 8; ++j) s += pr[c0 + j];
    float mu = wave_sum(s) * (1.f / 256.f);
    float v2 = 0.f;
#pragma unroll
    for (int j = 0; j < 8; ++j) { float d = pr[c0 + j] - mu; v2 += d * d; }
    float inv = rsqrtf(wave_sum(v2) * (1.f / 256.f) + LN_EPS);
    float* xr = X + (grow0 + row) * 256;
#pragma unroll
    for (int j = 0; j < 8; ++j) {
      int c = c0 + j;
      xr[c] = (pr[c] - mu) * inv * lns[c] + lnb[c];
    }
  }
}

// ------------------------------ small kernels ------------------------------

__global__ void glob1_kernel(const float* xg, const float* w1, const float* b1, float* tb) {
  int b = blockIdx.x, d = threadIdx.x;
  float s = b1[d];
#pragma unroll
  for (int k = 0; k < 16; ++k) s += xg[b * 16 + k] * w1[k * 256 + d];
  tb[b * 256 + d] = gelu_exact(s);
}
__global__ void glob2_kernel(const float* tb, const float* w2, const float* b2, float* g) {
  int b = blockIdx.x, d = threadIdx.x;
  float s = b2[d];
  for (int k = 0; k < 256; ++k) s += tb[b * 256 + k] * w2[k * 256 + d];
  g[b * 256 + d] = s;
}

__global__ void buildse_kernel(const float* cls_task, const float* globo, const float* empty,
                               const float* pos_emb, float* SE) {
  int blk = blockIdx.x, b = blk >> 5, row = blk & 31, d = threadIdx.x;
  float v;
  if (row < 7) v = cls_task[row * 256 + d];
  else if (row == 7) v = globo[b * 256 + d] + pos_emb[d];
  else if (row < 23) { int p = row - 8; v = empty[d] + pos_emb[(1 + p) * 256 + d]; }
  else v = 0.f;
  SE[((size_t)b * 32 + row) * 256 + d] = v;
}

__global__ void scattermod_kernel(const float* __restrict__ X, const int* m2e, const int* mpos,
                                  const float* pos_emb, float* SE) {
  int m = blockIdx.x, d = threadIdx.x;
  int e = m2e[m], p = mpos[m];
  SE[((size_t)e * 32 + 8 + p) * 256 + d] = X[(size_t)m * 112 * 256 + d] + pos_emb[(1 + p) * 256 + d];
}

__global__ void head_kernel(const float* __restrict__ SE, const float* __restrict__ hw,
                            const float* __restrict__ hb, float* __restrict__ out) {
  __shared__ float raw[16];
  int b = blockIdx.x, t = threadIdx.x, w = t >> 5, lane = t & 31;
  const int tok_of[16] = {0, 0, 0, 0, 1, 2, 3, 4, 5, 5, 5, 5, 6, 6, 6, 6};
#pragma unroll
  for (int i = 0; i < 2; ++i) {
    int col = w * 2 + i;
    int tok = tok_of[col];
    const float* e = SE + ((size_t)b * 32 + tok) * 256;
    float s = 0.f;
    for (int c = lane; c < 256; c += 32) s += e[c] * hw[c * 16 + col];
    s = wave_sum(s);
    if (lane == 0) raw[col] = s + hb[col];
  }
  __syncthreads();
  if (t < 16) {
    float x = raw[t];
    float res;
    if (t >= 9 && t < 12) {
      float n2 = raw[9] * raw[9] + raw[10] * raw[10] + raw[11] * raw[11];
      res = x / fmaxf(sqrtf(n2), 1e-12f);
    } else if (t >= 13) {
      float n2 = raw[13] * raw[13] + raw[14] * raw[14] + raw[15] * raw[15];
      res = x / fmaxf(sqrtf(n2), 1e-12f);
    } else {
      res = (x > 20.f) ? x : log1pf(__expf(x));
    }
    out[b * 16 + t] = res;
  }
}

// --------------------------------- driver ----------------------------------

extern "C" void kernel_launch(void* const* d_in, const int* in_sizes, int n_in,
                              void* d_out, int out_size, void* d_ws, size_t ws_size,
                              hipStream_t stream) {
  if (n_in < 33) return;
  const float* feats   = (const float*)d_in[0];
  const float* coords  = (const float*)d_in[1];
  const int*   bids    = (const int*)d_in[2];
  const int*   m2e     = (const int*)d_in[3];
  const int*   mpos    = (const int*)d_in[4];
  const float* xglob   = (const float*)d_in[5];
  const float* cls_mod = (const float*)d_in[6];
  const float* emptyv  = (const float*)d_in[7];
  const float* cls_task= (const float*)d_in[8];
  const float* pos_emb = (const float*)d_in[9];
  const float* maw     = (const float*)d_in[10];
  const float* mab     = (const float*)d_in[11];
  const float* mrel    = (const float*)d_in[12];
  const float* mlns    = (const float*)d_in[13];
  const float* mlnb    = (const float*)d_in[14];
  const float* mf_w1   = (const float*)d_in[15];
  const float* mf_b1   = (const float*)d_in[16];
  const float* mf_w2   = (const float*)d_in[17];
  const float* mf_b2   = (const float*)d_in[18];
  const float* gw1     = (const float*)d_in[19];
  const float* gb1     = (const float*)d_in[20];
  const float* gw2     = (const float*)d_in[21];
  const float* gb2     = (const float*)d_in[22];
  const float* eaw     = (const float*)d_in[23];
  const float* eab     = (const float*)d_in[24];
  const float* elns    = (const float*)d_in[25];
  const float* elnb    = (const float*)d_in[26];
  const float* ef_w1   = (const float*)d_in[27];
  const float* ef_b1   = (const float*)d_in[28];
  const float* ef_w2   = (const float*)d_in[29];
  const float* ef_b2   = (const float*)d_in[30];
  const float* head_w  = (const float*)d_in[31];
  const float* head_b  = (const float*)d_in[32];

  const int N = in_sizes[0] / 256;
  const int M = in_sizes[3];        // 960 modules
  const int B = in_sizes[5] / 16;   // 64 events

  char* ws = (char*)d_ws;
  size_t off = 0;
  auto take = [&](size_t bytes) -> void* {
    void* p = ws + off;
    off = (off + bytes + 255) & ~(size_t)255;
    return p;
  };
  _Float16* wqkvo_mod = (_Float16*)take((size_t)2 * 256 * 1024 * 2);
  _Float16* w1_mod    = (_Float16*)take((size_t)2 * 256 * 1024 * 2);
  _Float16* w2_mod    = (_Float16*)take((size_t)2 * 1024 * 256 * 2);
  _Float16* wqkvo_evt = (_Float16*)take((size_t)3 * 256 * 1024 * 2);
  _Float16* w1_evt    = (_Float16*)take((size_t)3 * 256 * 2048 * 2);
  _Float16* w2_evt    = (_Float16*)take((size_t)3 * 2048 * 256 * 2);
  float* X      = (float*)take((size_t)M * 112 * 256 * 4);
  float* pc     = (float*)take((size_t)M * 112 * 3 * 4);
  int*   counts = (int*)take((size_t)M * 4);
  int*   starts = (int*)take((size_t)M * 4);
  float* SE     = (float*)take((size_t)B * 32 * 256 * 4);
  float* tglob  = (float*)take((size_t)B * 256 * 4);
  float* gout   = (float*)take((size_t)B * 256 * 4);
  _Float16* QK  = (_Float16*)take((size_t)M * 112 * 512 * 2);
  _Float16* VT  = (_Float16*)take((size_t)M * 112 * 256 * 2);
  _Float16* OH  = (_Float16*)take((size_t)M * 112 * 256 * 2);
  (void)ws_size; (void)out_size;

  auto blocks = [](int n) { return (n + 255) / 256; };

  // weight packing (f32 -> fragment-tiled f16)
  packqkvo_kernel<<<blocks(2 * 256 * 1024), 256, 0, stream>>>(maw, wqkvo_mod, 2 * 256 * 1024);
  packqkvo_kernel<<<blocks(3 * 256 * 1024), 256, 0, stream>>>(eaw, wqkvo_evt, 3 * 256 * 1024);
  packw_kernel<<<blocks(2 * 256 * 1024), 256, 0, stream>>>(mf_w1, w1_mod, 256, 1024, 2 * 256 * 1024);
  packw_kernel<<<blocks(2 * 1024 * 256), 256, 0, stream>>>(mf_w2, w2_mod, 1024, 256, 2 * 1024 * 256);
  packw_kernel<<<blocks(3 * 256 * 2048), 256, 0, stream>>>(ef_w1, w1_evt, 256, 2048, 3 * 256 * 2048);
  packw_kernel<<<blocks(3 * 2048 * 256), 256, 0, stream>>>(ef_w2, w2_evt, 2048, 256, 3 * 2048 * 256);

  // sequence packing
  zero_kernel<<<4096, 256, 0, stream>>>(X, (size_t)M * 112 * 256);
  zero_kernel<<<1024, 256, 0, stream>>>(pc, (size_t)M * 112 * 3);
  zero_kernel<<<8, 256, 0, stream>>>((float*)counts, (size_t)M);
  count_kernel<<<blocks(N), 256, 0, stream>>>(bids, counts, N);
  starts_kernel<<<1, 64, 0, stream>>>(counts, starts, M);
  clsfill_kernel<<<M, 256, 0, stream>>>(cls_mod, X);
  scatter_kernel<<<N, 256, 0, stream>>>(feats, coords, bids, starts, X, pc);

  // module transformer (2 layers)
  for (int l = 0; l < 2; ++l) {
    qkv_kernel<<<dim3(M, 7, 3), 256, 0, stream>>>(
        X, wqkvo_mod + (size_t)l * 262144, mab + l * 1024, QK, VT, 112);
    attn_kernel<<<dim3(M, 8, 7), 256, 0, stream>>>(
        QK, VT, OH, pc, counts, mrel + l * 24, 112, 0);
    oproj_ln_kernel<<<dim3(M, 7), 256, 0, stream>>>(
        OH, wqkvo_mod + (size_t)l * 262144, mab + l * 1024 + 768,
        mlns + (l * 2 + 0) * 256, mlnb + (l * 2 + 0) * 256, X, 112);
    ffn_kernel<<<dim3(M, 7), 256, 0, stream>>>(
        X, w1_mod + (size_t)l * 262144, mf_b1 + l * 1024,
        w2_mod + (size_t)l * 262144, mf_b2 + l * 256,
        mlns + (l * 2 + 1) * 256, mlnb + (l * 2 + 1) * 256, 112, 1024, 0);
  }

  // event sequence assembly
  glob1_kernel<<<B, 256, 0, stream>>>(xglob, gw1, gb1, tglob);
  glob2_kernel<<<B, 256, 0, stream>>>(tglob, gw2, gb2, gout);
  buildse_kernel<<<B * 32, 256, 0, stream>>>(cls_task, gout, emptyv, pos_emb, SE);
  scattermod_kernel<<<M, 256, 0, stream>>>(X, m2e, mpos, pos_emb, SE);

  // event transformer (3 layers), padded rows key-masked at j >= 23
  for (int l = 0; l < 3; ++l) {
    qkv_kernel<<<dim3(B, 2, 3), 256, 0, stream>>>(
        SE, wqkvo_evt + (size_t)l * 262144, eab + l * 1024, QK, VT, 32);
    attn_kernel<<<dim3(B, 8, 2), 256, 0, stream>>>(
        QK, VT, OH, nullptr, nullptr, nullptr, 32, 23);
    oproj_ln_kernel<<<dim3(B, 2), 256, 0, stream>>>(
        OH, wqkvo_evt + (size_t)l * 262144, eab + l * 1024 + 768,
        elns + (l * 2 + 0) * 256, elnb + (l * 2 + 0) * 256, SE, 32);
    ffn_kernel<<<dim3(B, 2), 256, 0, stream>>>(
        SE, w1_evt + (size_t)l * 524288, ef_b1 + l * 2048,
        w2_evt + (size_t)l * 524288, ef_b2 + l * 256,
        elns + (l * 2 + 1) * 256, elnb + (l * 2 + 1) * 256, 32, 2048, 1);
  }

  // task heads
  head_kernel<<<B, 256, 0, stream>>>(SE, head_w, head_b, (float*)d_out);
}
